// VQ_71940702208348
// MI455X (gfx1250) — compile-verified, matrix operand order checked by reading, not run on hardware
//
#include <hip/hip_runtime.h>
#include <hip/hip_bf16.h>

// ---------------------------------------------------------------------------
// VQ nearest-codebook search, MI455X (gfx1250), wave32 + WMMA f16.
// dist argmin  ==  argmax_k ( 2*z.w_k - ||w_k||^2 )   (||z||^2 is row-const)
// ---------------------------------------------------------------------------

typedef __attribute__((ext_vector_type(16))) _Float16 v16h;
typedef __attribute__((ext_vector_type(8)))  _Float16 v8h;
typedef __attribute__((ext_vector_type(8)))  float    v8f;
typedef __attribute__((ext_vector_type(4)))  float    v4f;
typedef __attribute__((ext_vector_type(4)))  unsigned v4u;
typedef __attribute__((ext_vector_type(4)))  int      v4i;

#define AS1 __attribute__((address_space(1)))
#define AS3 __attribute__((address_space(3)))

#if defined(__has_builtin)
#  if __has_builtin(__builtin_amdgcn_global_load_async_to_lds_b128)
#    define HAS_ASYNC 1
#  endif
#endif
#ifndef HAS_ASYNC
#  define HAS_ASYNC 0
#endif

#if HAS_ASYNC
#  if __has_builtin(__builtin_amdgcn_s_wait_asynccnt)
#    define WAIT_ASYNC() __builtin_amdgcn_s_wait_asynccnt(0)
#  else
#    define WAIT_ASYNC() asm volatile("s_wait_asynccnt 0" ::: "memory")
#  endif
#else
#  define WAIT_ASYNC() ((void)0)
#endif

static constexpr int Nn = 65536;
static constexpr int Kk = 8192;
static constexpr int Dd = 256;
static constexpr int KT = 32;          // codebook rows per LDS chunk
static constexpr int DP = Dd + 8;      // padded LDS stride (halfs): 528B, bank-shifted
static constexpr int ROWS_WG = 128;    // 8 waves * 16 z-rows

// ---------------------------------------------------------------------------
// Prep: codebook fp32 -> fp16 in workspace, plus w2[k] = ||w_k||^2.
// One wave per codebook row.
// ---------------------------------------------------------------------------
__global__ __launch_bounds__(256) void vq_prep(const float* __restrict__ cb,
                                               _Float16* __restrict__ cbh,
                                               float* __restrict__ w2) {
  int row  = blockIdx.x * 8 + (threadIdx.x >> 5);
  int lane = threadIdx.x & 31;
  const float* src = cb + (size_t)row * Dd;
  _Float16*    dst = cbh + (size_t)row * Dd;
  float s = 0.f;
#pragma unroll
  for (int i = 0; i < 8; ++i) {
    float v = src[lane + i * 32];
    s += v * v;
    dst[lane + i * 32] = (_Float16)v;
  }
#pragma unroll
  for (int m = 16; m >= 1; m >>= 1) s += __shfl_xor(s, m, 32);
  if (lane == 0) w2[row] = s;
}

// ---------------------------------------------------------------------------
// Main: per workgroup 128 z-rows; scan all K codebook rows with f16 WMMA.
// ---------------------------------------------------------------------------
__global__ __launch_bounds__(256) void vq_main(const float* __restrict__ z,
                                               const float* __restrict__ cb,
                                               const _Float16* __restrict__ cbh,
                                               const float* __restrict__ w2g,
                                               float* __restrict__ zq,
                                               int* __restrict__ idx_out) {
  __shared__ _Float16 sB[2][KT * DP];   // codebook f16 tiles (double buffered)
  __shared__ float    sW2[2][KT];
  __shared__ int      sIdx[ROWS_WG];

  const int tid  = threadIdx.x;
  const int wave = tid >> 5;
  const int lane = tid & 31;
  const int hl   = lane & 15;           // half-lane (maps to M row / N col)
  const int hi   = lane >> 4;           // which 16-lane half
  const size_t blockRow0 = (size_t)blockIdx.x * ROWS_WG;
  const size_t rowBase   = blockRow0 + (size_t)wave * 16;

  // ---- Load & convert this wave's 16 z-rows into A fragments (once). ----
  // A layout (16-bit 16x32): lanes 0-15 hold M=0..15; per lane, halves 0-7 =
  // K=base+0..7 (lanes 0-15) / base+8..15 (lanes 16-31), halves 8-15 = +16.
  v16h a[8];
  {
    const float* zr = z + (rowBase + hl) * (size_t)Dd;
#pragma unroll
    for (int c = 0; c < 8; ++c) {
      int r0 = c * 32 + hi * 8;
      v4f f0 = *(const v4f*)(zr + r0);
      v4f f1 = *(const v4f*)(zr + r0 + 4);
      v4f f2 = *(const v4f*)(zr + r0 + 16);
      v4f f3 = *(const v4f*)(zr + r0 + 20);
#pragma unroll
      for (int j = 0; j < 4; ++j) {
        a[c][j]      = (_Float16)f0[j];
        a[c][4 + j]  = (_Float16)f1[j];
        a[c][8 + j]  = (_Float16)f2[j];
        a[c][12 + j] = (_Float16)f3[j];
      }
    }
  }

  float bmax[8];
  int   bidx[8];
#pragma unroll
  for (int r = 0; r < 8; ++r) { bmax[r] = -3.402823466e38f; bidx[r] = 0; }

  // ---- Chunk stager: 32 codebook rows (f16) -> LDS, async when available ----
  auto stage = [&](int buf, int kk0) {
#pragma unroll
    for (int i = 0; i < 4; ++i) {
      int linear = tid + i * 256;            // 16B-granule index, 0..1023
      int row = linear >> 5;                 // 0..31  (32 granules per row)
      int col = linear & 31;
      const _Float16* src = cbh + (size_t)(kk0 + row) * Dd + col * 8;
      _Float16*       dst = &sB[buf][row * DP + col * 8];
#if HAS_ASYNC
      __builtin_amdgcn_global_load_async_to_lds_b128((AS1 v4i*)src,
                                                     (AS3 v4i*)dst, 0, 0);
#else
      *(v4u*)dst = *(const v4u*)src;
#endif
    }
    if (tid < KT) sW2[buf][tid] = w2g[kk0 + tid];
    if (kk0 + 2 * KT < Kk)  // pull chunk+2 toward L2/L1
      __builtin_prefetch(cbh + (size_t)(kk0 + 2 * KT) * Dd + tid * 32, 0, 1);
  };

  stage(0, 0);
  WAIT_ASYNC();
  __syncthreads();

  const int NCHUNK = Kk / KT;   // 256
  for (int ch = 0; ch < NCHUNK; ++ch) {
    const int buf = ch & 1;
    const int kk0 = ch * KT;
    if (ch + 1 < NCHUNK) stage(buf ^ 1, kk0 + KT);

    // B operand: lane hl holds codebook row (kk0+n0+hl), d-runs mirror A's.
#pragma unroll
    for (int t = 0; t < 2; ++t) {
      const int n0 = t * 16;
      const _Float16* bp = &sB[buf][(n0 + hl) * DP];
      v8f acc = {};
#pragma unroll
      for (int c = 0; c < 8; ++c) {
        int r0 = c * 32 + hi * 8;
        v8h blo = *(const v8h*)(bp + r0);
        v8h bh2 = *(const v8h*)(bp + r0 + 16);
        v16h b;
#pragma unroll
        for (int j = 0; j < 8; ++j) { b[j] = blo[j]; b[j + 8] = bh2[j]; }
        acc = __builtin_amdgcn_wmma_f32_16x16x32_f16(
            false, a[c], false, b, (short)0, acc, false, false);
      }
      // acc VGPR r: lanes 0-15 -> (M=r, N=hl); lanes 16-31 -> (M=8+r, N=hl)
      float w    = sW2[buf][n0 + hl];
      int   kcol = kk0 + n0 + hl;
#pragma unroll
      for (int r = 0; r < 8; ++r) {
        float sc = 2.0f * acc[r] - w;
        if (sc > bmax[r]) { bmax[r] = sc; bidx[r] = kcol; }
      }
    }

    WAIT_ASYNC();
    __syncthreads();
  }

  // ---- Reduce (max, idx) across the 16 lanes of each half-wave. ----
#pragma unroll
  for (int r = 0; r < 8; ++r) {
    float v = bmax[r];
    int   bi = bidx[r];
#pragma unroll
    for (int m = 1; m < 16; m <<= 1) {
      float ov = __shfl_xor(v, m, 32);
      int   oi = __shfl_xor(bi, m, 32);
      if (ov > v || (ov == v && oi < bi)) { v = ov; bi = oi; }
    }
    if (hl == 0) sIdx[wave * 16 + hi * 8 + r] = bi;   // M = hi*8 + r
  }
  __syncthreads();

  // ---- Outputs: indices (int32 tail of d_out) + fp32 gather of z_q. ----
  if (tid < ROWS_WG) idx_out[blockRow0 + tid] = sIdx[tid];

  const int sub = tid & 7;                 // 8 threads per row
  for (int row = tid >> 3; row < ROWS_WG; row += 32) {
    int kbest = sIdx[row];
    const v4f* src = (const v4f*)(cb + (size_t)kbest * Dd);
    v4f*       dst = (v4f*)(zq + (blockRow0 + row) * (size_t)Dd);
#pragma unroll
    for (int i = 0; i < 8; ++i) dst[sub + i * 8] = src[sub + i * 8];
  }
}

// ---------------------------------------------------------------------------
extern "C" void kernel_launch(void* const* d_in, const int* in_sizes, int n_in,
                              void* d_out, int out_size, void* d_ws, size_t ws_size,
                              hipStream_t stream) {
  const float* z  = (const float*)d_in[0];   // (N, D) f32
  const float* cb = (const float*)d_in[1];   // (K, D) f32

  // workspace: [0, K*D*2) f16 codebook ; then K floats of w2
  _Float16* cbh = (_Float16*)d_ws;
  float*    w2  = (float*)((char*)d_ws + (size_t)Kk * Dd * sizeof(_Float16));

  // d_out: z_q (N*D f32) then indices (N int32) concatenated flat
  float* zq      = (float*)d_out;
  int*   idx_out = (int*)((float*)d_out + (size_t)Nn * Dd);

  vq_prep<<<dim3(Kk / 8), dim3(256), 0, stream>>>(cb, cbh, w2);
  vq_main<<<dim3(Nn / ROWS_WG), dim3(256), 0, stream>>>(z, cb, cbh, w2, zq, idx_out);
}
